// AttentionDecoder_7773890806017
// MI455X (gfx1250) — matrix operand order repaired and checked by reading, not compiled, not run on hardware
//
#include <hip/hip_runtime.h>
#include <math.h>

// ---------- dims ----------
#define VOCAB 32000
#define EDIM  512
#define HDIM  1024
#define LL    2
#define BB    64
#define TT    64
#define SS    128
#define VV    1024
#define KCAT  2048   // concat(input 1024, hidden 1024)

// TDM staging chunk: 64 rows x KC bf16 cols, padded row stride in LDS
#define KC    128
#define AROW  136    // 128 elems (256B) + 16B TDM pad = 272B = 136 shorts

// ---------- WMMA / vector types ----------
typedef __bf16  v16bf __attribute__((ext_vector_type(16)));
typedef float   v8f   __attribute__((ext_vector_type(8)));
typedef unsigned int v4u  __attribute__((ext_vector_type(4)));
typedef unsigned int u32x4 __attribute__((ext_vector_type(4)));
typedef int          i32x8 __attribute__((ext_vector_type(8)));
typedef int          i32x4 __attribute__((ext_vector_type(4)));

union FragAB { v16bf f; v4u u[2]; };

__device__ __forceinline__ unsigned short f2bf(float x) {
    union { float f; unsigned int u; } c; c.f = x;
    unsigned int u = c.u;
    unsigned int r = u + 0x7FFFu + ((u >> 16) & 1u);
    return (unsigned short)(r >> 16);
}

__device__ __forceinline__ float sigm(float x) { return 1.0f / (1.0f + __expf(-x)); }

__device__ __forceinline__ unsigned int ldsoff(const void* p) {
    // flat shared address: low 32 bits are the LDS byte offset (aperture rules)
    return (unsigned int)(unsigned long long)(uintptr_t)p;
}

// ---------- Tensor Data Mover: DMA [64 x KC] bf16 tile (row stride lda elems) into LDS ----------
// D# per CDNA5 ISA ch.8: group0 = count/lds_addr/global_addr/type, group1 = sizes/strides/pad.
// LDS rows padded by 16B every 256B (pad_interval=5 -> 2^5*8=256B, pad_amount=3 -> 4 DWORDs).
__device__ __forceinline__ void tdm_load_a(const unsigned short* gsrc, int lda_elems,
                                           unsigned int lds_byte_addr) {
    unsigned long long ga = (unsigned long long)(uintptr_t)gsrc;
    u32x4 g0;
    g0[0] = 1u;                                        // count=1 (valid), user mode
    g0[1] = lds_byte_addr;                             // lds_addr
    g0[2] = (unsigned int)ga;                          // global_addr[31:0]
    g0[3] = (unsigned int)((ga >> 32) & 0x01FFFFFFu)   // global_addr[56:32]
          | (2u << 30);                                // type=2 ("image")
    const unsigned int td0 = KC, td1 = 64;             // tensor dims == tile dims (no OOB)
    unsigned long long s0 = (unsigned long long)(unsigned int)lda_elems;
    i32x8 g1;
    g1[0] = (int)((1u << 16)                           // data_size=1 -> 2 bytes
                | (1u << 20)                           // pad_enable
                | (5u << 22)                           // pad_interval: every 256B
                | (3u << 25));                         // pad_amount: 4 DWORDs (16B)
    g1[1] = (int)((td0 & 0xFFFFu) << 16);              // [63:48] tensor_dim0 lo
    g1[2] = (int)((td0 >> 16) | ((td1 & 0xFFFFu) << 16)); // dim0 hi | dim1 lo
    g1[3] = (int)(((td1 >> 16) & 0xFFFFu) | ((unsigned)KC << 16)); // dim1 hi | tile_dim0
    g1[4] = (int)(64u);                                // tile_dim1=64, tile_dim2=0
    g1[5] = (int)(unsigned int)(s0 & 0xFFFFFFFFu);     // tensor_dim0_stride lo
    g1[6] = (int)(unsigned int)((s0 >> 32) & 0xFFFFu); // stride hi | stride1 lo(0)
    g1[7] = 0;
    i32x4 gz4 = {0, 0, 0, 0};                          // groups 2/3: 2D tensor
    i32x8 gz8 = {0, 0, 0, 0, 0, 0, 0, 0};
    __builtin_amdgcn_tensor_load_to_lds(g0, g1, gz4, gz4, gz8, 0);
}

// ---------- setup kernels ----------
__global__ void k_cvt_wcat(const float* __restrict__ W_ih, const float* __restrict__ W_hh,
                           unsigned short* __restrict__ Wcat) {
    size_t i = (size_t)blockIdx.x * blockDim.x + threadIdx.x;  // 2*4096*2048
    size_t k = i & 2047;
    size_t n = (i >> 11) & 4095;
    size_t l = i >> 23;
    float v = (k < 1024) ? W_ih[(l * 4096 + n) * 1024 + k]
                         : W_hh[(l * 4096 + n) * 1024 + (k - 1024)];
    Wcat[i] = f2bf(v);
}

__global__ void k_cvt(const float* __restrict__ src, unsigned short* __restrict__ dst, int n) {
    int i = blockIdx.x * blockDim.x + threadIdx.x;
    if (i < n) dst[i] = f2bf(src[i]);
}

__global__ void k_init(const float* __restrict__ b_ih, const float* __restrict__ b_hh,
                       float* __restrict__ biasc, float* __restrict__ cst,
                       unsigned short* __restrict__ h0new, unsigned short* __restrict__ h1new,
                       unsigned short* __restrict__ outbf) {
    int i = blockIdx.x * blockDim.x + threadIdx.x;
    if (i < 2 * 4096)      biasc[i] = b_ih[i] + b_hh[i];
    if (i < 2 * BB * HDIM) cst[i] = 0.0f;
    if (i < BB * HDIM)     { h0new[i] = 0; h1new[i] = 0; }
    if (i < BB * EDIM)     outbf[i] = 0;
}

// ---------- per-step prep ----------
__global__ void k_prep(const int* __restrict__ tgt_t, const float* __restrict__ emb,
                       const unsigned short* __restrict__ outbf,
                       const unsigned short* __restrict__ h0new,
                       const unsigned short* __restrict__ h1new,
                       unsigned short* __restrict__ act0, unsigned short* __restrict__ act1) {
    int b = blockIdx.x, tid = threadIdx.x;
    int tok = tgt_t[b];
    for (int e = tid; e < EDIM; e += 256) act0[b * KCAT + e] = f2bf(emb[(size_t)tok * EDIM + e]);
    for (int e = tid; e < EDIM; e += 256) act0[b * KCAT + EDIM + e] = outbf[b * EDIM + e];
    for (int e = tid; e < HDIM; e += 256) act0[b * KCAT + 1024 + e] = h0new[b * HDIM + e];
    for (int e = tid; e < HDIM; e += 256) act1[b * KCAT + 1024 + e] = h1new[b * HDIM + e];
}

// ---------- fused LSTM layer: TDM-staged A, bf16 WMMA, LDS gate exchange, cell update ----------
__global__ void __launch_bounds__(128)
k_lstm(const unsigned short* __restrict__ act,    // [64, 2048] bf16
       const unsigned short* __restrict__ Wcat,   // [4096, 2048] bf16 (i,f,g,o)
       const float* __restrict__ biasc,           // [4096]
       float* __restrict__ cstate,                // [64, 1024] fp32
       unsigned short* __restrict__ hdst0,        // [64,1024] bf16
       unsigned short* __restrict__ hdst1, int hd1_stride, int hd1_off) {
    __shared__ __align__(16) unsigned short sbuf[2][64 * AROW];  // 2 x 17KB staged A
    __shared__ float lds_g[4][64][16];                           // 16 KB gate exchange
    const int wave = threadIdx.x >> 5;             // = gate index
    const int lane = threadIdx.x & 31;
    const int half = lane >> 4;
    const int l16  = lane & 15;
    const int jc   = blockIdx.x * 16;

    v8f acc[4] = {};
    const unsigned short* wp = Wcat + (size_t)(wave * HDIM + jc + l16) * KCAT + half * 16;

    if (wave == 0) tdm_load_a(act, KCAT, ldsoff(&sbuf[0][0]));
    const int nch = KCAT / KC;  // 16
    for (int c = 0; c < nch; ++c) {
        if (wave == 0) {
            if (c + 1 < nch) {
                tdm_load_a(act + (c + 1) * KC, KCAT, ldsoff(&sbuf[(c + 1) & 1][0]));
                __builtin_amdgcn_s_wait_tensorcnt(1);   // chunk c landed, c+1 in flight
            } else {
                __builtin_amdgcn_s_wait_tensorcnt(0);
            }
        }
        __syncthreads();
        const unsigned short* sb = &sbuf[c & 1][0];
        __builtin_prefetch(wp + (c + 1) * KC, 0, 1);    // next weight chunk into caches
#pragma unroll
        for (int kk = 0; kk < KC; kk += 32) {
            FragAB bf;
            bf.u[0] = *(const v4u*)(wp + c * KC + kk);
            bf.u[1] = *(const v4u*)(wp + c * KC + kk + 8);
            FragAB af[4];                               // batch all A ds-loads first
#pragma unroll
            for (int mt = 0; mt < 4; ++mt) {
                const unsigned short* ap = sb + (mt * 16 + l16) * AROW + kk + half * 16;
                af[mt].u[0] = *(const v4u*)(ap);
                af[mt].u[1] = *(const v4u*)(ap + 8);
            }
#pragma unroll
            for (int mt = 0; mt < 4; ++mt)
                acc[mt] = __builtin_amdgcn_wmma_f32_16x16x32_bf16(
                    false, af[mt].f, false, bf.f, (short)0, acc[mt], false, false);
        }
        __syncthreads();
    }

#pragma unroll
    for (int mt = 0; mt < 4; ++mt)
#pragma unroll
        for (int r = 0; r < 8; ++r)
            lds_g[wave][mt * 16 + half * 8 + r][l16] = acc[mt][r];
    __syncthreads();

    const int col = jc + l16;
    const float bi = biasc[col];
    const float bf_ = biasc[1024 + col];
    const float bg = biasc[2048 + col];
    const float bo = biasc[3072 + col];
#pragma unroll
    for (int r = 0; r < 8; ++r) {
        int row = wave * 16 + half * 8 + r;
        float gi = lds_g[0][row][l16] + bi;
        float gf = lds_g[1][row][l16] + bf_;
        float gg = lds_g[2][row][l16] + bg;
        float go = lds_g[3][row][l16] + bo;
        float co = cstate[row * HDIM + col];
        float cn = sigm(gf) * co + sigm(gi) * tanhf(gg);
        float hn = sigm(go) * tanhf(cn);
        cstate[row * HDIM + col] = cn;
        unsigned short hb = f2bf(hn);
        hdst0[row * HDIM + col] = hb;
        hdst1[row * hd1_stride + hd1_off + col] = hb;
    }
}

// ---------- generic [64 x N] = A[64 x K] * W[N x K]^T + bias, TDM-staged A ----------
__global__ void __launch_bounds__(128)
k_gemm(const unsigned short* __restrict__ A, int lda,
       const unsigned short* __restrict__ W, int K,
       const float* __restrict__ bias,
       float* __restrict__ C, int ldc,
       unsigned short* __restrict__ Cbf, int ldcbf,
       int do_tanh) {
    __shared__ __align__(16) unsigned short sbuf[2][64 * AROW];
    const int wave = threadIdx.x >> 5;
    const int lane = threadIdx.x & 31;
    const int half = lane >> 4;
    const int l16  = lane & 15;
    const int n0   = blockIdx.x * 64 + wave * 16;

    v8f acc[4] = {};
    const unsigned short* wp = W + (size_t)(n0 + l16) * K + half * 16;

    if (wave == 0) tdm_load_a(A, lda, ldsoff(&sbuf[0][0]));
    const int nch = K / KC;
    for (int c = 0; c < nch; ++c) {
        if (wave == 0) {
            if (c + 1 < nch) {
                tdm_load_a(A + (c + 1) * KC, lda, ldsoff(&sbuf[(c + 1) & 1][0]));
                __builtin_amdgcn_s_wait_tensorcnt(1);
            } else {
                __builtin_amdgcn_s_wait_tensorcnt(0);
            }
        }
        __syncthreads();
        const unsigned short* sb = &sbuf[c & 1][0];
        __builtin_prefetch(wp + (c + 1) * KC, 0, 1);
#pragma unroll
        for (int kk = 0; kk < KC; kk += 32) {
            FragAB bf;
            bf.u[0] = *(const v4u*)(wp + c * KC + kk);
            bf.u[1] = *(const v4u*)(wp + c * KC + kk + 8);
            FragAB af[4];
#pragma unroll
            for (int mt = 0; mt < 4; ++mt) {
                const unsigned short* ap = sb + (mt * 16 + l16) * AROW + kk + half * 16;
                af[mt].u[0] = *(const v4u*)(ap);
                af[mt].u[1] = *(const v4u*)(ap + 8);
            }
#pragma unroll
            for (int mt = 0; mt < 4; ++mt)
                acc[mt] = __builtin_amdgcn_wmma_f32_16x16x32_bf16(
                    false, af[mt].f, false, bf.f, (short)0, acc[mt], false, false);
        }
        __syncthreads();
    }

    const float bv = bias ? bias[n0 + l16] : 0.0f;
#pragma unroll
    for (int mt = 0; mt < 4; ++mt)
#pragma unroll
        for (int r = 0; r < 8; ++r) {
            int row = mt * 16 + half * 8 + r;
            float v = acc[mt][r] + bv;
            if (do_tanh) v = tanhf(v);
            if (C)   C[row * ldc + n0 + l16] = v;
            if (Cbf) Cbf[row * ldcbf + n0 + l16] = f2bf(v);
        }
}

// ---------- attention: scores + softmax + attended ----------
__global__ void __launch_bounds__(128)
k_attn(const float* __restrict__ q,        // [64,1024]
       const float* __restrict__ henc,     // [64,128,1024]
       unsigned short* __restrict__ attcat) {
    __shared__ float qs[VV];
    __shared__ float red[SS];
    __shared__ float aw[SS];
    const int b = blockIdx.x, tid = threadIdx.x;  // 128 threads = S
    for (int i = tid; i < VV; i += 128) qs[i] = q[b * VV + i];
    __syncthreads();

    const float4* h4 = reinterpret_cast<const float4*>(henc + ((size_t)b * SS + tid) * VV);
    float dot = 0.0f;
    for (int v = 0; v < VV / 4; ++v) {
        float4 hv = h4[v];
        dot += qs[4 * v] * hv.x + qs[4 * v + 1] * hv.y + qs[4 * v + 2] * hv.z + qs[4 * v + 3] * hv.w;
    }
    red[tid] = dot;
    __syncthreads();
    for (int off = 64; off; off >>= 1) {
        if (tid < off) red[tid] = fmaxf(red[tid], red[tid + off]);
        __syncthreads();
    }
    float mx = red[0];
    __syncthreads();
    float e = __expf(dot - mx);
    red[tid] = e;
    __syncthreads();
    for (int off = 64; off; off >>= 1) {
        if (tid < off) red[tid] += red[tid + off];
        __syncthreads();
    }
    aw[tid] = e / red[0];
    __syncthreads();

    for (int j = 0; j < 8; ++j) {
        int v = tid + 128 * j;
        const float* hc = henc + (size_t)b * SS * VV + v;
        float a = 0.0f;
#pragma unroll 4
        for (int s = 0; s < SS; ++s) a += aw[s] * hc[(size_t)s * VV];
        attcat[b * KCAT + v] = f2bf(a);
    }
}

// ---------- host orchestration ----------
extern "C" void kernel_launch(void* const* d_in, const int* in_sizes, int n_in,
                              void* d_out, int out_size, void* d_ws, size_t ws_size,
                              hipStream_t stream) {
    const int*   tgt  = (const int*)d_in[0];
    const float* henc = (const float*)d_in[1];
    const float* emb  = (const float*)d_in[2];
    const float* W_ih = (const float*)d_in[5];
    const float* W_hh = (const float*)d_in[6];
    const float* b_ih = (const float*)d_in[7];
    const float* b_hh = (const float*)d_in[8];
    const float* W1   = (const float*)d_in[9];
    const float* b1   = (const float*)d_in[10];
    const float* W2   = (const float*)d_in[11];
    const float* b2   = (const float*)d_in[12];
    float* out = (float*)d_out;

    char* ws = (char*)d_ws;
    size_t off = 0;
    auto alloc = [&](size_t bytes) -> void* {
        void* p = ws + off;
        off = (off + bytes + 255) & ~(size_t)255;
        return p;
    };
    unsigned short* Wcat  = (unsigned short*)alloc((size_t)LL * 4096 * KCAT * 2);
    unsigned short* W1b   = (unsigned short*)alloc((size_t)VV * HDIM * 2);
    unsigned short* W2b   = (unsigned short*)alloc((size_t)EDIM * KCAT * 2);
    float*          biasc = (float*)alloc((size_t)LL * 4096 * 4);
    float*          cst   = (float*)alloc((size_t)LL * BB * HDIM * 4);
    unsigned short* h0new = (unsigned short*)alloc((size_t)BB * HDIM * 2);
    unsigned short* h1new = (unsigned short*)alloc((size_t)BB * HDIM * 2);
    unsigned short* outbf = (unsigned short*)alloc((size_t)BB * EDIM * 2);
    unsigned short* act0  = (unsigned short*)alloc((size_t)BB * KCAT * 2);
    unsigned short* act1  = (unsigned short*)alloc((size_t)BB * KCAT * 2);
    unsigned short* attct = (unsigned short*)alloc((size_t)BB * KCAT * 2);
    float*          qbuf  = (float*)alloc((size_t)BB * VV * 4);

    k_cvt_wcat<<<65536, 256, 0, stream>>>(W_ih, W_hh, Wcat);
    k_cvt<<<(VV * HDIM) / 256, 256, 0, stream>>>(W1, W1b, VV * HDIM);
    k_cvt<<<(EDIM * KCAT) / 256, 256, 0, stream>>>(W2, W2b, EDIM * KCAT);
    k_init<<<512, 256, 0, stream>>>(b_ih, b_hh, biasc, cst, h0new, h1new, outbf);

    for (int t = 0; t < TT; ++t) {
        k_prep<<<BB, 256, 0, stream>>>(tgt + t * BB, emb, outbf, h0new, h1new, act0, act1);
        k_lstm<<<64, 128, 0, stream>>>(act0, Wcat, biasc, cst, h0new, act1, KCAT, 0);
        k_lstm<<<64, 128, 0, stream>>>(act1, Wcat + (size_t)4096 * KCAT, biasc + 4096,
                                       cst + BB * HDIM, h1new, attct, KCAT, 1024);
        k_gemm<<<VV / 64, 128, 0, stream>>>(h1new, HDIM, W1b, HDIM, b1,
                                            qbuf, VV, (unsigned short*)nullptr, 0, 0);
        k_attn<<<BB, 128, 0, stream>>>(qbuf, henc, attct);
        k_gemm<<<EDIM / 64, 128, 0, stream>>>(attct, KCAT, W2b, KCAT, b2,
                                              out + (size_t)t * EDIM, TT * EDIM,
                                              outbf, EDIM, 1);
    }
}
